// PNNStack_64819646431359
// MI455X (gfx1250) — matrix operand order repaired and checked by reading, not compiled, not run on hardware
//
#include <hip/hip_runtime.h>
#include <math.h>

typedef __attribute__((ext_vector_type(16))) _Float16 v16h;
typedef __attribute__((ext_vector_type(8)))  float    v8f;

#define NNODES 50000
#define NEDGES 800000
#define FDIM   64
#define NLAYER 3
#define NGRAPH 50
#define NHEADS 3
#define KPRE   128      // 2F
#define KPOST  1088     // 17F
#define NF     (NNODES * FDIM)

// ---------------- float atomic min/max via sign-aware integer atomics --------
__device__ __forceinline__ void atomicMinF(float* addr, float v) {
    if (v >= 0.0f) atomicMin((int*)addr, __float_as_int(v));
    else           atomicMax((unsigned int*)addr, __float_as_uint(v));
}
__device__ __forceinline__ void atomicMaxF(float* addr, float v) {
    if (v >= 0.0f) atomicMax((int*)addr, __float_as_int(v));
    else           atomicMin((unsigned int*)addr, __float_as_uint(v));
}

// ---------------- small utility kernels --------------------------------------
__global__ void cvt_f16_kernel(const float* __restrict__ in, _Float16* __restrict__ out, int n) {
    int i = blockIdx.x * blockDim.x + threadIdx.x;
    if (i < n) out[i] = (_Float16)in[i];
}

__global__ void init_x_kernel(const float* __restrict__ x, float* __restrict__ xcur,
                              _Float16* __restrict__ xh, int n) {
    int i = blockIdx.x * blockDim.x + threadIdx.x;
    if (i < n) { float v = x[i]; xcur[i] = v; xh[i] = (_Float16)v; }
}

__global__ void calc_scal_kernel(const float* __restrict__ dh, int nbins, float* __restrict__ scal) {
    if (blockIdx.x == 0 && threadIdx.x == 0) {
        float tot = 0.f, lin = 0.f, lg = 0.f;
        for (int i = 0; i < nbins; ++i) {
            float d = dh[i];
            tot += d; lin += (float)i * d; lg += logf((float)i + 1.0f) * d;
        }
        scal[0] = lg / tot;   // avg_log
        scal[1] = lin / tot;  // avg_lin
    }
}

__global__ void init_layer_kernel(float* cnt, float* sum, float* sum2,
                                  float* mn, float* mx, float* bnsum, float* bnsum2) {
    int i = blockIdx.x * blockDim.x + threadIdx.x;
    if (i < NF) {
        sum[i] = 0.f; sum2[i] = 0.f;
        mn[i] = __int_as_float(0x7F800000);   // +inf
        mx[i] = __int_as_float(0xFF800000);   // -inf
    }
    if (i < NNODES) cnt[i] = 0.f;
    if (i < FDIM) { bnsum[i] = 0.f; bnsum2[i] = 0.f; }
}

__global__ void edge_count_kernel(const int* __restrict__ dst, float* __restrict__ cnt, int E) {
    int e = blockIdx.x * blockDim.x + threadIdx.x;
    if (e < E) atomicAdd(&cnt[dst[e]], 1.0f);
}

// ---------------- edge GEMM: h = [x[dst]|x[src]] @ preW + preb, fused segment atomics
// 128 threads = 4 waves; wave handles 16 edges x 64 outputs.
// Gathered rows are staged global->LDS with GLOBAL_LOAD_ASYNC_TO_LDS_B128 (ASYNCcnt).
__global__ __launch_bounds__(128) void edge_gemm_kernel(
    const _Float16* __restrict__ xh, const int* __restrict__ src, const int* __restrict__ dst,
    const _Float16* __restrict__ preWh, const float* __restrict__ preb,
    float* __restrict__ sum, float* __restrict__ sum2,
    float* __restrict__ mn, float* __restrict__ mx, int E)
{
    __shared__ _Float16 ldsW[4 * 4 * 32 * 16];                 // packed B fragments, 16KB
    __shared__ __align__(16) _Float16 ldsA[4][16][KPRE];       // per-wave A rows, 16KB
    __shared__ int ldsD[4][16];

    const int t = threadIdx.x;
    const int wave = t >> 5, lane = t & 31;
    const int m = lane & 15, hi = lane >> 4;
    const int e0 = blockIdx.x * 64 + wave * 16;

    // async-stage A rows: lane (m, hi) moves 128B of x[dst] (hi=0) / x[src] (hi=1)
    // straight into LDS, no VGPR round-trip.
    {
        int e  = e0 + m;
        int ec = (e < E) ? e : 0;
        int node = hi ? src[ec] : dst[ec];
        const char* gp = (const char*)(xh + (size_t)node * FDIM);
        unsigned lo = (unsigned)(uintptr_t)&ldsA[wave][m][hi * 64];
        #pragma unroll
        for (int q = 0; q < 8; ++q) {
            asm volatile("global_load_async_to_lds_b128 %0, %1, off"
                         :: "v"(lo + q * 16), "v"(gp + (size_t)(q * 16))
                         : "memory");
        }
        if (hi == 0) ldsD[wave][m] = (e < E) ? dst[e] : -1;
    }

    // stage preW fragments, packed so compute reads are contiguous 32B per lane
    for (int q = t; q < 8192; q += 128) {
        int j  = q & 15;
        int ln = (q >> 4) & 31;
        int nt = (q >> 9) & 3;
        int kt = q >> 11;
        int k  = kt * 32 + ((ln & 16) ? 16 : 0) + j;   // B: lanes 0-15 K=0..15, 16-31 K=16..31
        int n  = nt * 16 + (ln & 15);
        ldsW[q] = preWh[k * FDIM + n];
    }

    asm volatile("s_wait_asynccnt 0x0" ::: "memory");  // our A rows landed in LDS
    __syncthreads();

    v8f acc[4] = {};
    #pragma unroll
    for (int kt = 0; kt < 4; ++kt) {
        v16h a;
        #pragma unroll
        for (int p = 0; p < 8; ++p) {     // A layout: vgpr p -> K pair, half-wave K offset 16
            int kb = hi * 16 + p * 2;
            int gk = kt * 32 + kb;
            a[2 * p]     = ldsA[wave][m][gk];
            a[2 * p + 1] = ldsA[wave][m][gk + 1];
        }
        #pragma unroll
        for (int nt = 0; nt < 4; ++nt) {
            v16h b;
            const _Float16* bp = &ldsW[((kt * 4 + nt) * 32 + lane) * 16];
            #pragma unroll
            for (int j = 0; j < 16; ++j) b[j] = bp[j];
            acc[nt] = __builtin_amdgcn_wmma_f32_16x16x32_f16(
                false, a, false, b, (short)0, acc[nt], false, false);
        }
    }

    // epilogue: C layout lanes 0-15 -> M=v, lanes 16-31 -> M=8+v; N = nt*16 + (lane&15)
    int dsts[8];
    #pragma unroll
    for (int v = 0; v < 8; ++v) dsts[v] = ldsD[wave][v + hi * 8];

    #pragma unroll
    for (int nt = 0; nt < 4; ++nt) {
        int n = nt * 16 + m;
        float pb = preb[n];
        #pragma unroll
        for (int v = 0; v < 8; ++v) {
            int d = dsts[v];
            if (d >= 0) {
                float hv = acc[nt][v] + pb;
                size_t idx = (size_t)d * FDIM + n;
                atomicAdd(&sum[idx], hv);
                atomicAdd(&sum2[idx], hv * hv);
                atomicMinF(&mn[idx], hv);
                atomicMaxF(&mx[idx], hv);
            }
        }
    }
}

// ---------------- build [N, 17F] f16 A-matrix for the post-MLP ---------------
__global__ void build_aggr_kernel(
    const float* __restrict__ xcur, const float* __restrict__ cnt,
    const float* __restrict__ sum, const float* __restrict__ sum2,
    const float* __restrict__ mn, const float* __restrict__ mx,
    const float* __restrict__ scal, _Float16* __restrict__ aggr17)
{
    int idx = blockIdx.x * blockDim.x + threadIdx.x;
    if (idx >= NF) return;
    int node = idx >> 6, f = idx & 63;
    float c    = cnt[node];
    float deg  = fmaxf(c, 1.0f);
    float inv  = 1.0f / deg;
    float mean = sum[idx] * inv;
    float var  = sum2[idx] * inv - mean * mean;
    var = fmaxf(var, 0.0f);
    float sd = sqrtf(var + 1e-5f);
    bool has = (c > 0.0f);
    float vmn = has ? mn[idx] : 0.0f;
    float vmx = has ? mx[idx] : 0.0f;
    float avg_log = scal[0], avg_lin = scal[1];
    float ld = logf(deg + 1.0f);
    float s1 = ld / avg_log, s2 = avg_log / ld, s3 = deg / avg_lin;

    _Float16* r = aggr17 + (size_t)node * KPOST;
    r[f] = (_Float16)xcur[idx];
    float a0 = mean, a1 = vmn, a2 = vmx, a3 = sd;
    int b = 64 + f;
    r[b] = (_Float16)a0; r[b+64] = (_Float16)a1; r[b+128] = (_Float16)a2; r[b+192] = (_Float16)a3;
    b += 256;
    r[b] = (_Float16)(a0*s1); r[b+64] = (_Float16)(a1*s1); r[b+128] = (_Float16)(a2*s1); r[b+192] = (_Float16)(a3*s1);
    b += 256;
    r[b] = (_Float16)(a0*s2); r[b+64] = (_Float16)(a1*s2); r[b+128] = (_Float16)(a2*s2); r[b+192] = (_Float16)(a3*s2);
    b += 256;
    r[b] = (_Float16)(a0*s3); r[b+64] = (_Float16)(a1*s3); r[b+128] = (_Float16)(a2*s3); r[b+192] = (_Float16)(a3*s3);
}

// ---------------- post GEMM [N,1088]@[1088,64] + fused [64,64] linear --------
// 128 threads = 4 waves; wave handles 16 nodes x 64 outputs.
__global__ __launch_bounds__(128) void post_gemm_kernel(
    const _Float16* __restrict__ aggr17, const _Float16* __restrict__ postWh,
    const float* __restrict__ postb, const _Float16* __restrict__ linWh,
    const float* __restrict__ linb, float* __restrict__ y,
    float* __restrict__ bnsum, float* __restrict__ bnsum2)
{
    __shared__ _Float16 ldsB[4 * 32 * 16];                   // one K-tile of postW, 4KB
    __shared__ _Float16 ldsL[2 * 4 * 32 * 16];               // linW fragments, 8KB
    __shared__ __align__(16) _Float16 ldsT[4][16][FDIM];     // per-wave mid tile, 8KB
    __shared__ float ldsS[FDIM], ldsS2[FDIM];

    const int t = threadIdx.x;
    const int wave = t >> 5, lane = t & 31;
    const int m = lane & 15, hi = lane >> 4;
    const int nbase = blockIdx.x * 64 + wave * 16;
    int row = nbase + m; if (row >= NNODES) row = NNODES - 1;
    const _Float16* arow = aggr17 + (size_t)row * KPOST;

    for (int q = t; q < 4096; q += 128) {                    // stage linW fragments once
        int j  = q & 15;
        int ln = (q >> 4) & 31;
        int nt = (q >> 9) & 3;
        int kt = q >> 11;
        int k  = kt * 32 + ((ln & 16) ? 16 : 0) + j;
        int n  = nt * 16 + (ln & 15);
        ldsL[q] = linWh[k * FDIM + n];
    }
    if (t < FDIM) { ldsS[t] = 0.f; ldsS2[t] = 0.f; }

    v8f acc[4] = {};
    for (int kt = 0; kt < 34; ++kt) {
        __syncthreads();
        for (int q = t; q < 2048; q += 128) {                // stage postW K-tile
            int j  = q & 15;
            int ln = (q >> 4) & 31;
            int nt = q >> 9;
            int k  = kt * 32 + ((ln & 16) ? 16 : 0) + j;
            int n  = nt * 16 + (ln & 15);
            ldsB[q] = postWh[(size_t)k * FDIM + n];
        }
        if (kt < 33)                                          // hide next A-tile latency
            __builtin_prefetch(arow + (kt + 1) * 32, 0, 3);
        __syncthreads();
        v16h a;
        #pragma unroll
        for (int p = 0; p < 8; ++p) {
            int kb = hi * 16 + p * 2;
            int gk = kt * 32 + kb;
            a[2 * p]     = arow[gk];
            a[2 * p + 1] = arow[gk + 1];
        }
        #pragma unroll
        for (int nt = 0; nt < 4; ++nt) {
            v16h b;
            const _Float16* bp = &ldsB[(nt * 32 + lane) * 16];
            #pragma unroll
            for (int j = 0; j < 16; ++j) b[j] = bp[j];
            acc[nt] = __builtin_amdgcn_wmma_f32_16x16x32_f16(
                false, a, false, b, (short)0, acc[nt], false, false);
        }
    }

    // t = acc + postb  ->  LDS f16 tile (wave-local)
    #pragma unroll
    for (int nt = 0; nt < 4; ++nt) {
        int n = nt * 16 + m;
        float pb = postb[n];
        #pragma unroll
        for (int v = 0; v < 8; ++v)
            ldsT[wave][v + hi * 8][n] = (_Float16)(acc[nt][v] + pb);
    }
    __syncthreads();

    // second GEMM: t @ linW, K = 64 (2 K-tiles)
    v8f acc2[4] = {};
    #pragma unroll
    for (int kt = 0; kt < 2; ++kt) {
        v16h a;
        #pragma unroll
        for (int p = 0; p < 8; ++p) {
            int kb = hi * 16 + p * 2;
            int gk = kt * 32 + kb;
            a[2 * p]     = ldsT[wave][m][gk];
            a[2 * p + 1] = ldsT[wave][m][gk + 1];
        }
        #pragma unroll
        for (int nt = 0; nt < 4; ++nt) {
            v16h b;
            const _Float16* bp = &ldsL[((kt * 4 + nt) * 32 + lane) * 16];
            #pragma unroll
            for (int j = 0; j < 16; ++j) b[j] = bp[j];
            acc2[nt] = __builtin_amdgcn_wmma_f32_16x16x32_f16(
                false, a, false, b, (short)0, acc2[nt], false, false);
        }
    }

    // epilogue: y out + BN partial sums (block-reduced in LDS, then 1 atomic/feature)
    #pragma unroll
    for (int nt = 0; nt < 4; ++nt) {
        int n = nt * 16 + m;
        float lb = linb[n];
        #pragma unroll
        for (int v = 0; v < 8; ++v) {
            int node = nbase + v + hi * 8;
            if (node < NNODES) {
                float yv = acc2[nt][v] + lb;
                y[(size_t)node * FDIM + n] = yv;
                atomicAdd(&ldsS[n], yv);
                atomicAdd(&ldsS2[n], yv * yv);
            }
        }
    }
    __syncthreads();
    if (t < FDIM) {
        atomicAdd(&bnsum[t], ldsS[t]);
        atomicAdd(&bnsum2[t], ldsS2[t]);
    }
}

// ---------------- BN (training-mode, biased var) + ReLU ----------------------
__global__ void bn_relu_kernel(const float* __restrict__ y,
                               const float* __restrict__ bnsum, const float* __restrict__ bnsum2,
                               const float* __restrict__ gamma, const float* __restrict__ beta,
                               float* __restrict__ xcur, _Float16* __restrict__ xh)
{
    int idx = blockIdx.x * blockDim.x + threadIdx.x;
    if (idx >= NF) return;
    int f = idx & 63;
    const float invN = 1.0f / (float)NNODES;
    float mu = bnsum[f] * invN;
    float var = bnsum2[f] * invN - mu * mu;
    float v = gamma[f] * (y[idx] - mu) * rsqrtf(var + 1e-5f) + beta[f];
    v = fmaxf(v, 0.0f);
    xcur[idx] = v;
    xh[idx] = (_Float16)v;
}

// ---------------- graph pooling ----------------------------------------------
__global__ void pool_init_kernel(float* gpool, float* gcnt) {
    int i = blockIdx.x * blockDim.x + threadIdx.x;
    if (i < NGRAPH * FDIM) gpool[i] = 0.f;
    if (i < NGRAPH) gcnt[i] = 0.f;
}

__global__ void pool_kernel(const float* __restrict__ xcur, const int* __restrict__ batch,
                            float* __restrict__ gpool, float* __restrict__ gcnt)
{
    int idx = blockIdx.x * blockDim.x + threadIdx.x;
    if (idx >= NF) return;
    int node = idx >> 6, f = idx & 63;
    int b = batch[node];
    atomicAdd(&gpool[b * FDIM + f], xcur[idx]);
    if (f == 0) atomicAdd(&gcnt[b], 1.0f);
}

// ---------------- shared MLP + 3 head MLPs (tiny) ----------------------------
__global__ void heads_kernel(const float* __restrict__ gpool, const float* __restrict__ gcnt,
                             const float* __restrict__ sharedW, const float* __restrict__ sharedb,
                             const float* __restrict__ hW1, const float* __restrict__ hb1,
                             const float* __restrict__ hW2, const float* __restrict__ hb2,
                             const float* __restrict__ hW3, const float* __restrict__ hb3,
                             float* __restrict__ out)
{
    int g = threadIdx.x;
    if (g >= NGRAPH) return;
    float gin[FDIM], gs[FDIM];
    float c = fmaxf(gcnt[g], 1.0f);
    for (int k = 0; k < FDIM; ++k) gin[k] = gpool[g * FDIM + k] / c;
    for (int f = 0; f < FDIM; ++f) {
        float a = sharedb[f];
        for (int k = 0; k < FDIM; ++k) a += gin[k] * sharedW[k * FDIM + f];
        gs[f] = fmaxf(a, 0.0f);
    }
    for (int h = 0; h < NHEADS; ++h) {
        float h1[50];
        for (int j = 0; j < 50; ++j) {
            float a = hb1[h * 50 + j];
            for (int k = 0; k < FDIM; ++k) a += gs[k] * hW1[((size_t)h * FDIM + k) * 50 + j];
            h1[j] = fmaxf(a, 0.0f);
        }
        float h2[25];
        for (int j = 0; j < 25; ++j) {
            float a = hb2[h * 25 + j];
            for (int k = 0; k < 50; ++k) a += h1[k] * hW2[((size_t)h * 50 + k) * 25 + j];
            h2[j] = fmaxf(a, 0.0f);
        }
        float o = hb3[h];
        for (int k = 0; k < 25; ++k) o += h2[k] * hW3[h * 25 + k];
        out[g * NHEADS + h] = o;   // [G, H]
    }
}

// =============================================================================
extern "C" void kernel_launch(void* const* d_in, const int* in_sizes, int n_in,
                              void* d_out, int out_size, void* d_ws, size_t ws_size,
                              hipStream_t stream)
{
    const float* x        = (const float*)d_in[0];
    const int*   eidx     = (const int*)  d_in[1];
    const int*   batch    = (const int*)  d_in[2];
    const float* deg_hist = (const float*)d_in[3];
    const float* preW     = (const float*)d_in[4];
    const float* preb     = (const float*)d_in[5];
    const float* postW    = (const float*)d_in[6];
    const float* postb    = (const float*)d_in[7];
    const float* linW     = (const float*)d_in[8];
    const float* linb     = (const float*)d_in[9];
    const float* bn_g     = (const float*)d_in[10];
    const float* bn_b     = (const float*)d_in[11];
    const float* sharedW  = (const float*)d_in[12];
    const float* sharedb  = (const float*)d_in[13];
    const float* hW1      = (const float*)d_in[14];
    const float* hb1      = (const float*)d_in[15];
    const float* hW2      = (const float*)d_in[16];
    const float* hb2      = (const float*)d_in[17];
    const float* hW3      = (const float*)d_in[18];
    const float* hb3      = (const float*)d_in[19];
    float* out = (float*)d_out;

    const int* src = eidx;
    const int* dst = eidx + NEDGES;
    const int nbins = in_sizes[3];

    // ---- workspace carve-up (256B aligned) ----
    char* w = (char*)d_ws;
    size_t off = 0;
    auto take = [&](size_t bytes) -> void* {
        void* p = w + off;
        off = (off + bytes + 255) & ~(size_t)255;
        return p;
    };
    float*    xcur   = (float*)   take((size_t)NF * 4);
    _Float16* xh     = (_Float16*)take((size_t)NF * 2);
    float*    cnt    = (float*)   take((size_t)NNODES * 4);
    float*    sum    = (float*)   take((size_t)NF * 4);
    float*    sum2   = (float*)   take((size_t)NF * 4);
    float*    mn     = (float*)   take((size_t)NF * 4);
    float*    mx     = (float*)   take((size_t)NF * 4);
    _Float16* aggr17 = (_Float16*)take((size_t)NNODES * KPOST * 2);
    float*    ybuf   = (float*)   take((size_t)NF * 4);
    float*    bnsum  = (float*)   take(FDIM * 4);
    float*    bnsum2 = (float*)   take(FDIM * 4);
    float*    scal   = (float*)   take(2 * 4);
    float*    gpool  = (float*)   take(NGRAPH * FDIM * 4);
    float*    gcnt   = (float*)   take(NGRAPH * 4);
    _Float16* preWh  = (_Float16*)take((size_t)NLAYER * KPRE  * FDIM * 2);
    _Float16* postWh = (_Float16*)take((size_t)NLAYER * KPOST * FDIM * 2);
    _Float16* linWh  = (_Float16*)take((size_t)NLAYER * FDIM  * FDIM * 2);
    (void)ws_size; (void)n_in; (void)out_size;

    const int TB = 256;
    const int gridNF = (NF + TB - 1) / TB;

    // ---- one-time prep ----
    {
        int n = NLAYER * KPRE * FDIM;
        cvt_f16_kernel<<<(n + TB - 1) / TB, TB, 0, stream>>>(preW, preWh, n);
        n = NLAYER * KPOST * FDIM;
        cvt_f16_kernel<<<(n + TB - 1) / TB, TB, 0, stream>>>(postW, postWh, n);
        n = NLAYER * FDIM * FDIM;
        cvt_f16_kernel<<<(n + TB - 1) / TB, TB, 0, stream>>>(linW, linWh, n);
    }
    init_x_kernel<<<gridNF, TB, 0, stream>>>(x, xcur, xh, NF);
    calc_scal_kernel<<<1, 32, 0, stream>>>(deg_hist, nbins, scal);

    // ---- 3 PNA layers ----
    for (int l = 0; l < NLAYER; ++l) {
        init_layer_kernel<<<gridNF, TB, 0, stream>>>(cnt, sum, sum2, mn, mx, bnsum, bnsum2);
        edge_count_kernel<<<(NEDGES + TB - 1) / TB, TB, 0, stream>>>(dst, cnt, NEDGES);
        edge_gemm_kernel<<<(NEDGES + 63) / 64, 128, 0, stream>>>(
            xh, src, dst,
            preWh + (size_t)l * KPRE * FDIM, preb + l * FDIM,
            sum, sum2, mn, mx, NEDGES);
        build_aggr_kernel<<<gridNF, TB, 0, stream>>>(xcur, cnt, sum, sum2, mn, mx, scal, aggr17);
        post_gemm_kernel<<<(NNODES + 63) / 64, 128, 0, stream>>>(
            aggr17, postWh + (size_t)l * KPOST * FDIM, postb + l * FDIM,
            linWh + (size_t)l * FDIM * FDIM, linb + l * FDIM,
            ybuf, bnsum, bnsum2);
        bn_relu_kernel<<<gridNF, TB, 0, stream>>>(
            ybuf, bnsum, bnsum2, bn_g + l * FDIM, bn_b + l * FDIM, xcur, xh);
    }

    // ---- pooling + heads ----
    pool_init_kernel<<<(NGRAPH * FDIM + TB - 1) / TB, TB, 0, stream>>>(gpool, gcnt);
    pool_kernel<<<gridNF, TB, 0, stream>>>(xcur, batch, gpool, gcnt);
    heads_kernel<<<1, 64, 0, stream>>>(gpool, gcnt, sharedW, sharedb,
                                       hW1, hb1, hW2, hb2, hW3, hb3, out);
}
